// VAEEncoder_56032143343742
// MI455X (gfx1250) — matrix operand order repaired and checked by reading, not compiled, not run on hardware
//
#include <hip/hip_runtime.h>
#include <hip/hip_bf16.h>

// ---------------------------------------------------------------------------
// LSTM stack (L=8, H=512, B=256, T=256, E=128) + MLP head for MI455X (gfx1250)
// bf16 WMMA (v_wmma_f32_16x16x32_bf16), fp32 accumulation.
//  1) convert weights fp32->bf16, transpose x [B,T,E]->[T,B,E] bf16
//  2) per layer: parallel WMMA GEMM xp = xs @ W_ih^T + b, written PRE-SWIZZLED
//     into WMMA C-fragment order [t][wg][mtile][nt][lane][v] (bf16) so the
//     recurrent kernel's accumulator init is one global_load_b128 per frag.
//  3) per layer: persistent 128-WG recurrent kernel. Batch is split into 8
//     independent groups of 32 rows (recurrence is sequential in t only);
//     each group has 16 hidden-slice WGs + its own step barrier. W_hh slice
//     (128KB bf16) resident in LDS; c-state resident in VGPRs; h ping-pong
//     through L2. All 16 h A-fragments are preloaded per step (one clause)
//     so a timestep pays ~one L2 latency instead of 16.
//  4) tiny MLP head.
// ---------------------------------------------------------------------------

#define HD   512
#define NL   8
#define ED   128
#define BB   256
#define TT   256
#define G4H  2048   // 4*HD

typedef __attribute__((ext_vector_type(16))) __bf16        v16bf;
typedef __attribute__((ext_vector_type(8)))  float         v8f;
typedef __attribute__((ext_vector_type(4)))  unsigned int  u32x4;

struct FragBits { u32x4 lo; u32x4 hi; };

static __device__ __forceinline__ unsigned short f2bf(float f) {
  unsigned int u = __builtin_bit_cast(unsigned int, f);
  u += 0x7fffu + ((u >> 16) & 1u);                 // round-to-nearest-even
  return (unsigned short)(u >> 16);
}
static __device__ __forceinline__ float bf2f(unsigned short h) {
  unsigned int u = ((unsigned int)h) << 16;
  return __builtin_bit_cast(float, u);
}
static __device__ __forceinline__ float bf_lo(unsigned int d) {
  return __builtin_bit_cast(float, d << 16);
}
static __device__ __forceinline__ float bf_hi(unsigned int d) {
  return __builtin_bit_cast(float, d & 0xffff0000u);
}
// Build a 16x bf16 WMMA fragment from two 16-byte chunks.
static __device__ __forceinline__ v16bf frag_from(const unsigned short* p0,
                                                  const unsigned short* p1) {
  FragBits fb;
  fb.lo = *(const u32x4*)p0;
  fb.hi = *(const u32x4*)p1;
  return __builtin_bit_cast(v16bf, fb);
}
static __device__ __forceinline__ float sigmoidf_(float x) {
  return 1.0f / (1.0f + __expf(-x));
}
static __device__ __forceinline__ float tanhf_(float x) {
  return 2.0f * sigmoidf_(2.0f * x) - 1.0f;
}

// --------------------------- conversion kernels ----------------------------
__global__ __launch_bounds__(256) void f32_to_bf16_kernel(
    const float* __restrict__ in, unsigned short* __restrict__ out, int n) {
  for (int i = blockIdx.x * 256 + threadIdx.x; i < n; i += gridDim.x * 256)
    out[i] = f2bf(in[i]);
}

// x: [B,T,E] f32  ->  x0: [T,B,E] bf16
__global__ __launch_bounds__(256) void x_transpose_kernel(
    const float* __restrict__ x, unsigned short* __restrict__ x0) {
  int i = blockIdx.x * 256 + threadIdx.x;            // index over output
  if (i >= BB * TT * ED) return;
  int e  = i & (ED - 1);
  int tb = i >> 7;                                   // log2(ED)=7
  int b  = tb & (BB - 1);
  int t  = tb >> 8;                                  // log2(BB)=8
  x0[i] = f2bf(x[((size_t)b * TT + t) * ED + e]);
}

// ------------------------ input-projection WMMA GEMM -----------------------
// xp_swz[(((t*16+wg)*16+mtile)*8+nt)*32*8 + lane*8 + v] = xs@W^T + b (bf16),
// i.e. the exact per-lane C-fragment layout the recurrent kernel consumes.
// Block = 256 thr (8 waves) -> 128x128 tile. Wave = 64(M) x 32(N).
__global__ __launch_bounds__(256) void proj_gemm_kernel(
    const unsigned short* __restrict__ xs, const unsigned short* __restrict__ Wt,
    const float* __restrict__ bias, unsigned short* __restrict__ xp, int K) {
  const int lane = threadIdx.x & 31;
  const int w    = threadIdx.x >> 5;
  const int wrow = w >> 2, wcol = w & 3;
  const int mbase = blockIdx.y * 128 + wrow * 64;
  const int nbase = blockIdx.x * 128 + wcol * 32;
  const int l15 = lane & 15, lhi = lane >> 4;        // 0/1

  v8f acc[4][2];
#pragma unroll
  for (int mt = 0; mt < 4; ++mt)
#pragma unroll
    for (int nt = 0; nt < 2; ++nt)
      acc[mt][nt] = v8f{0.f, 0.f, 0.f, 0.f, 0.f, 0.f, 0.f, 0.f};

  const int kiters = K >> 5;
  for (int kt = 0; kt < kiters; ++kt) {
    const int kA = kt * 32 + (lhi ? 8 : 0);
    v16bf a[4];
#pragma unroll
    for (int mt = 0; mt < 4; ++mt) {
      const unsigned short* pa = xs + (size_t)(mbase + mt * 16 + l15) * K + kA;
      a[mt] = frag_from(pa, pa + 16);
    }
    v16bf bfr[2];
#pragma unroll
    for (int nt = 0; nt < 2; ++nt) {
      const unsigned short* pb =
          Wt + (size_t)(nbase + nt * 16 + l15) * K + kt * 32 + (lhi ? 8 : 0);
      bfr[nt] = frag_from(pb, pb + 16);
    }
#pragma unroll
    for (int mt = 0; mt < 4; ++mt)
#pragma unroll
      for (int nt = 0; nt < 2; ++nt)
        acc[mt][nt] = __builtin_amdgcn_wmma_f32_16x16x32_bf16(
            false, a[mt], false, bfr[nt], (short)0, acc[mt][nt], false, false);
  }

  // Epilogue: +bias, pack 8 bf16 -> one 16B store per fragment (swizzled).
#pragma unroll
  for (int nt = 0; nt < 2; ++nt) {
    const int col = nbase + nt * 16 + l15;
    const float bv = bias[col];
    const int wg_  = (col & 511) >> 5;
    const int nt_  = ((col >> 9) << 1) | ((col >> 4) & 1);
#pragma unroll
    for (int mt = 0; mt < 4; ++mt) {
      const int row0 = mbase + mt * 16 + lhi * 8;    // row of v=0
      const int t_   = row0 >> 8;
      const int b0   = row0 & 255;
      const int mtile = b0 >> 4;
      const int lane_ = ((b0 >> 3) & 1) * 16 + l15;
      u32x4 q;
#pragma unroll
      for (int d = 0; d < 4; ++d) {
        const unsigned int lo = f2bf(acc[mt][nt][2 * d] + bv);
        const unsigned int hi = f2bf(acc[mt][nt][2 * d + 1] + bv);
        q[d] = lo | (hi << 16);
      }
      const size_t base =
          ((((size_t)t_ * 16 + wg_) * 16 + mtile) * 8 + nt_) * 256 +
          (size_t)lane_ * 8;
      *(u32x4*)(xp + base) = q;
    }
  }
}

// ----------------------------- recurrent kernel ----------------------------
// 128 WGs x 128 thr. blockIdx = bg*16 + wg: batch group bg (32 rows), hidden
// slice wg (32 units). 4 waves: (mt = w>>1) row-tile, (p = w&1) gate-half.
// Each wave: 4 gate fragments + 1 c-state fragment, 64 WMMA / step.
__global__ __launch_bounds__(128) void lstm_recurrent_kernel(
    const unsigned short* __restrict__ xp,   // swizzled bf16 (see proj)
    const float* __restrict__ whh,           // [2048, 512] fp32
    unsigned short* __restrict__ hs_out,     // [T*B, 512] bf16
    unsigned short* __restrict__ hbuf0, unsigned short* __restrict__ hbuf1,
    unsigned short* __restrict__ finals_l,   // [B, 512] bf16
    unsigned int* __restrict__ sync_layer) { // 8 counters (one per group)
  extern __shared__ unsigned short smem_w[];            // [128][512] bf16
  const int tid = threadIdx.x;
  const int bg  = blockIdx.x >> 4;                      // 0..7
  const int wg  = blockIdx.x & 15;                      // 0..15
  const int j0  = wg * 32;

  // Stage + convert W_hh slice: LDS row n = g*32+jl  <-  W_hh[g*512 + j0 + jl]
  for (int idx = tid; idx < 128 * 512; idx += 128) {
    const int n = idx >> 9, k = idx & 511;
    const int g = n >> 5, jl = n & 31;
    smem_w[idx] = f2bf(whh[(size_t)(g * 512 + j0 + jl) * HD + k]);
  }
  __syncthreads();

  const int lane = tid & 31, w = tid >> 5;              // w: 0..3
  const int mt = w >> 1, p = w & 1;
  const int l15 = lane & 15, lhi = lane >> 4;
  const int rowbase = bg * 32 + mt * 16;
  const int mtile   = bg * 2 + mt;                      // 0..15
  const int arow    = rowbase + l15;
  const int hu      = j0 + p * 16 + l15;
  unsigned int* cnt = sync_layer + bg;

  float c[8];
#pragma unroll
  for (int v = 0; v < 8; ++v) c[v] = 0.f;

  for (int t = 0; t < TT; ++t) {
    const unsigned short* hprev = (t & 1) ? hbuf0 : hbuf1;  // h_{t-1}
    unsigned short*       hcur  = (t & 1) ? hbuf1 : hbuf0;  // h_t

    // accumulator init = precomputed input projection (+bias), one b128 each
    v8f acc[4];
#pragma unroll
    for (int g = 0; g < 4; ++g) {
      const int nt = g * 2 + p;
      const size_t base =
          ((((size_t)t * 16 + wg) * 16 + mtile) * 8 + nt) * 256 +
          (size_t)lane * 8;
      const u32x4 q = *(const u32x4*)(xp + base);
#pragma unroll
      for (int d = 0; d < 4; ++d) {
        acc[g][2 * d]     = bf_lo(q[d]);
        acc[g][2 * d + 1] = bf_hi(q[d]);
      }
    }

    if (t > 0) {                                    // h_{-1} == 0
      // Preload the whole K=512 slice of h rows: 16 A-frags in one clause,
      // then a fully unrolled 64-WMMA chain against LDS B-frags.
      v16bf afr[16];
#pragma unroll
      for (int kt = 0; kt < 16; ++kt) {
        const unsigned short* pa =
            hprev + (size_t)arow * HD + kt * 32 + (lhi ? 8 : 0);
        afr[kt] = frag_from(pa, pa + 16);
      }
#pragma unroll
      for (int kt = 0; kt < 16; ++kt) {
#pragma unroll
        for (int g = 0; g < 4; ++g) {
          const unsigned short* pb =
              smem_w + (size_t)(g * 32 + p * 16 + l15) * HD + kt * 32 +
              (lhi ? 8 : 0);
          const v16bf bfr = frag_from(pb, pb + 16);
          acc[g] = __builtin_amdgcn_wmma_f32_16x16x32_bf16(
              false, afr[kt], false, bfr, (short)0, acc[g], false, false);
        }
      }
    }

    // elementwise LSTM cell update (i,f,g,o in matching lanes)
#pragma unroll
    for (int v = 0; v < 8; ++v) {
      const float iv = sigmoidf_(acc[0][v]);
      const float fv = sigmoidf_(acc[1][v]);
      const float gv = tanhf_(acc[2][v]);
      const float ov = sigmoidf_(acc[3][v]);
      const float cn = fv * c[v] + iv * gv;
      c[v] = cn;
      const unsigned short hb = f2bf(ov * tanhf_(cn));
      const int row = rowbase + lhi * 8 + v;
      hcur[(size_t)row * HD + hu] = hb;
      hs_out[((size_t)t * BB + row) * HD + hu] = hb;
      if (t == TT - 1) finals_l[(size_t)row * HD + hu] = hb;
    }

    if (t < TT - 1)   // warm caches for next step's projection fragments
      __builtin_prefetch(
          xp + ((((size_t)(t + 1) * 16 + wg) * 16 + mtile) * 8) * 256, 0, 1);

    // per-group step barrier: release stores, arrive, spin.
    __threadfence();
    __syncthreads();
    __builtin_amdgcn_s_cluster_barrier();   // real sync if clustered, else NOP
    if (tid == 0) {
      __hip_atomic_fetch_add(cnt, 1u, __ATOMIC_RELEASE,
                             __HIP_MEMORY_SCOPE_AGENT);
      const unsigned int target = (unsigned int)(t + 1) * 16u;
      while (__hip_atomic_load(cnt, __ATOMIC_ACQUIRE,
                               __HIP_MEMORY_SCOPE_AGENT) < target)
        __builtin_amdgcn_s_sleep(2);
    }
    __syncthreads();
  }
}

// -------------------------------- MLP head ---------------------------------
__global__ __launch_bounds__(256) void head_ff1_kernel(
    const unsigned short* __restrict__ finals, const float* __restrict__ w1,
    const float* __restrict__ b1, float* __restrict__ tmp) {
  const int i = blockIdx.x * 256 + threadIdx.x;      // over 2048*128
  if (i >= (NL * BB) * (HD / 4)) return;
  const int j = i & 127, m = i >> 7;                 // m = l*256 + b
  const unsigned short* hr = finals + (size_t)m * HD;
  const float* wr = w1 + (size_t)j * HD;
  float s = b1[j];
  for (int k = 0; k < HD; ++k) s += bf2f(hr[k]) * wr[k];
  tmp[(size_t)m * (HD / 4) + j] = s * sigmoidf_(s);  // silu
}

__global__ __launch_bounds__(256) void head_ff2_kernel(
    const float* __restrict__ tmp, const float* __restrict__ w2,
    const float* __restrict__ b2, float* __restrict__ out) {
  const int i = blockIdx.x * 256 + threadIdx.x;      // over 2048*32
  if (i >= (NL * BB) * (HD / 16)) return;
  const int jj = i & 31, m = i >> 5;
  const int l = m >> 8, b = m & (BB - 1);
  const float* tr = tmp + (size_t)m * (HD / 4);
  const float* wr = w2 + (size_t)jj * (HD / 4);
  float s = b2[jj];
  for (int k = 0; k < HD / 4; ++k) s += tr[k] * wr[k];
  out[(size_t)b * (NL * (HD / 16)) + l * (HD / 16) + jj] = s;
}

// ------------------------------- launcher ----------------------------------
extern "C" void kernel_launch(void* const* d_in, const int* in_sizes, int n_in,
                              void* d_out, int out_size, void* d_ws,
                              size_t ws_size, hipStream_t stream) {
  (void)in_sizes; (void)n_in; (void)out_size; (void)ws_size;
  const float* x     = (const float*)d_in[0];
  const float* W_ih0 = (const float*)d_in[1];
  const float* W_ih  = (const float*)d_in[2];
  const float* W_hh  = (const float*)d_in[3];
  const float* bcomb = (const float*)d_in[4];
  const float* ff1_w = (const float*)d_in[5];
  const float* ff1_b = (const float*)d_in[6];
  const float* ff2_w = (const float*)d_in[7];
  const float* ff2_b = (const float*)d_in[8];
  float* out = (float*)d_out;

  unsigned char* ws = (unsigned char*)d_ws;
  size_t off = 0;
  auto carve = [&](size_t bytes) {
    size_t o = off;
    off = (off + bytes + 255) & ~(size_t)255;
    return o;
  };
  const size_t SYNC   = carve(4096);
  const size_t WIH0   = carve((size_t)G4H * ED * 2);             // bf16
  const size_t WIH    = carve((size_t)(NL - 1) * G4H * HD * 2);  // bf16
  const size_t X0     = carve((size_t)TT * BB * ED * 2);         // bf16
  const size_t XSB0   = carve((size_t)TT * BB * HD * 2);         // bf16
  const size_t XSB1   = carve((size_t)TT * BB * HD * 2);         // bf16
  const size_t XP     = carve((size_t)TT * BB * G4H * 2);        // bf16
  const size_t HB0    = carve((size_t)BB * HD * 2);              // bf16
  const size_t HB1    = carve((size_t)BB * HD * 2);              // bf16
  const size_t FIN    = carve((size_t)NL * BB * HD * 2);         // bf16
  const size_t TMP1   = carve((size_t)NL * BB * (HD / 4) * 4);   // f32

  unsigned int*   sync_cnt = (unsigned int*)(ws + SYNC);
  unsigned short* wih0_bf  = (unsigned short*)(ws + WIH0);
  unsigned short* wih_bf   = (unsigned short*)(ws + WIH);
  unsigned short* x0_bf    = (unsigned short*)(ws + X0);
  unsigned short* xsb[2]   = {(unsigned short*)(ws + XSB0),
                              (unsigned short*)(ws + XSB1)};
  unsigned short* xp_bf    = (unsigned short*)(ws + XP);
  unsigned short* hb0      = (unsigned short*)(ws + HB0);
  unsigned short* hb1      = (unsigned short*)(ws + HB1);
  unsigned short* fin_bf   = (unsigned short*)(ws + FIN);
  float*          tmp1     = (float*)(ws + TMP1);

  hipMemsetAsync(ws + SYNC, 0, 4096, stream);   // per-layer/group counters

  f32_to_bf16_kernel<<<1024, 256, 0, stream>>>(W_ih0, wih0_bf, G4H * ED);
  f32_to_bf16_kernel<<<8192, 256, 0, stream>>>(W_ih, wih_bf,
                                               (NL - 1) * G4H * HD);
  x_transpose_kernel<<<(BB * TT * ED + 255) / 256, 256, 0, stream>>>(x, x0_bf);

  for (int l = 0; l < NL; ++l) {
    const int K = (l == 0) ? ED : HD;
    const unsigned short* xs_in =
        (l == 0) ? x0_bf : xsb[(l - 1) & 1];
    const unsigned short* Wbf =
        (l == 0) ? wih0_bf : (wih_bf + (size_t)(l - 1) * G4H * HD);
    unsigned short* hs_out = xsb[l & 1];

    // xp = xs @ W_ih^T + b   (M=65536, N=2048), swizzled bf16 output
    proj_gemm_kernel<<<dim3(G4H / 128, (TT * BB) / 128), 256, 0, stream>>>(
        xs_in, Wbf, bcomb + (size_t)l * G4H, xp_bf, K);

    // persistent recurrent scan over T (8 batch groups x 16 hidden slices)
    lstm_recurrent_kernel<<<128, 128, 128 * 512 * 2, stream>>>(
        xp_bf, W_hh + (size_t)l * G4H * HD, hs_out, hb0, hb1,
        fin_bf + (size_t)l * BB * HD, sync_cnt + l * 8);
  }

  head_ff1_kernel<<<(NL * BB * (HD / 4) + 255) / 256, 256, 0, stream>>>(
      fin_bf, ff1_w, ff1_b, tmp1);
  head_ff2_kernel<<<(NL * BB * (HD / 16) + 255) / 256, 256, 0, stream>>>(
      tmp1, ff2_w, ff2_b, out);
}
